// CantorAttention_46523085750371
// MI455X (gfx1250) — matrix operand order repaired and checked by reading, not compiled, not run on hardware
//
#include <hip/hip_runtime.h>
#include <math.h>

#define DIM       1024
#define HEADS     16
#define HEAD_DIM  64
#define K_NEIGH   64
#define ATTN_SCALE 0.125f   // 1/sqrt(64)

typedef __attribute__((ext_vector_type(16))) __bf16 v16bf;
typedef __attribute__((ext_vector_type(8)))  __bf16 v8bf;
typedef __attribute__((ext_vector_type(8)))  float  v8f;

__device__ __forceinline__ unsigned short f32_to_bf16_rte(float f) {
  unsigned int u = __float_as_uint(f);
  unsigned int r = 0x7FFFu + ((u >> 16) & 1u);
  u += r;
  return (unsigned short)(u >> 16);
}

// ---------------------------------------------------------------------------
// Elementwise f32 -> bf16
// ---------------------------------------------------------------------------
__global__ void cvt_f32_bf16(const float* __restrict__ in,
                             unsigned short* __restrict__ out, int count) {
  int i = blockIdx.x * blockDim.x + threadIdx.x;
  if (i < count) out[i] = f32_to_bf16_rte(in[i]);
}

// ---------------------------------------------------------------------------
// Transpose (K x N) f32 -> (N x K) bf16, so GEMM B-columns become contiguous
// ---------------------------------------------------------------------------
__global__ void transpose_cvt_f32_bf16(const float* __restrict__ in,
                                       unsigned short* __restrict__ out,
                                       int K, int N) {
  int i = blockIdx.x * blockDim.x + threadIdx.x;
  if (i >= K * N) return;
  int k = i / N;
  int ncol = i - k * N;
  out[(size_t)ncol * K + k] = f32_to_bf16_rte(in[i]);
}

// ---------------------------------------------------------------------------
// Brute-force top-64 nearest neighbors in 1-D cantor coordinate space.
// One thread per query; sorted insertion list (stable: ties keep lower index).
// ---------------------------------------------------------------------------
__global__ void routes_kernel(const float* __restrict__ c,
                              int* __restrict__ routes, int n) {
  int q = blockIdx.x * blockDim.x + threadIdx.x;
  if (q >= n) return;
  float dist[K_NEIGH];
  int   idx [K_NEIGH];
  int count = 0;
  float cq = c[q];
  for (int j = 0; j < n; ++j) {
    float d = fabsf(cq - c[j]);
    if (count < K_NEIGH) {
      int p = count++;
      while (p > 0 && dist[p - 1] > d) { dist[p] = dist[p - 1]; idx[p] = idx[p - 1]; --p; }
      dist[p] = d; idx[p] = j;
    } else if (d < dist[K_NEIGH - 1]) {
      int p = K_NEIGH - 1;
      while (p > 0 && dist[p - 1] > d) { dist[p] = dist[p - 1]; idx[p] = idx[p - 1]; --p; }
      dist[p] = d; idx[p] = j;
    }
  }
  for (int j = 0; j < K_NEIGH; ++j) routes[q * K_NEIGH + j] = idx[j];
}

// ---------------------------------------------------------------------------
// bf16 WMMA GEMM:  C(MxN, f32) = A(MxK, bf16) * BT(NxK, bf16)^T + bias
//
// One wave computes a 16x64 output strip (4 accumulators; A fragment reused
// across 4 v_wmma_f32_16x16x32_bf16 per K-step -> ~12.8 flops / L2 byte).
//
// Software pipeline: K-loop unrolled x2 with explicit ping-pong fragment
// buffers (A-set / B-set). Each half-iteration issues the NEXT slab's 5x2
// b128 loads into the idle buffer set, then drains 4 WMMAs from the live set
// behind a partial s_wait_loadcnt. No buffer-rotation register moves, no
// WMMA->VALU hazard nops. Requires K % 64 == 0 (K = 1024 in both uses).
//
// Operand packing used here (wave32, contiguous k order in A and B):
//   A 16x32 : lane L -> row = L&15 ; half = L>>4
//             v16bf[0..7]  = A[row][kb + half*16 + 0..7]
//             v16bf[8..15] = A[row][kb + half*16 + 8..15]
//   B 32x16 : lane L -> col = L&15 ; lanes 0-15 hold K=0..15, 16-31 K=16..31
//             v16bf[0..15] = B[kb + (L>>4)*16 + 0..15][col]  (contiguous in BT)
//   C/D f32 : VGPR g -> row = tm + g + 8*(L>>4), col = tn + (L&15)
// ---------------------------------------------------------------------------
#define NT 4  // 16x16 tiles per wave in N (16x64 strip)

__device__ __forceinline__ v16bf pack_frag(const unsigned short* p0,
                                           const unsigned short* p1) {
  v8bf x = *reinterpret_cast<const v8bf*>(p0);
  v8bf y = *reinterpret_cast<const v8bf*>(p1);
  v16bf r;
#pragma unroll
  for (int i = 0; i < 8; ++i) { r[i] = x[i]; r[i + 8] = y[i]; }
  return r;
}

__global__ __launch_bounds__(256) void gemm_bf16_wmma(
    const unsigned short* __restrict__ A,   // M x K row-major bf16
    const unsigned short* __restrict__ BT,  // N x K row-major bf16 (B^T)
    const float* __restrict__ bias,         // length N (may be null)
    float* __restrict__ C,                  // M x N row-major f32
    int M, int N, int K) {
  int gwave = (int)((blockIdx.x * blockDim.x + threadIdx.x) >> 5);
  int lane  = threadIdx.x & 31;
  int stripsN = N >> 6;                 // 64-wide strips (N % 64 == 0 here)
  int stripsTotal = (M >> 4) * stripsN;
  if (gwave >= stripsTotal) return;

  int tm = (gwave / stripsN) << 4;
  int tn = (gwave % stripsN) << 6;
  int r    = lane & 15;
  int hsel = lane >> 4;

  int aoff = hsel << 4;   // 0 or 16
  int boff = hsel << 4;   // 0 or 16

  const unsigned short* arow = A + (size_t)(tm + r) * K + aoff;
  const unsigned short* brow[NT];
#pragma unroll
  for (int t = 0; t < NT; ++t)
    brow[t] = BT + (size_t)(tn + (t << 4) + r) * K + boff;

  v8f acc[NT];
#pragma unroll
  for (int t = 0; t < NT; ++t) acc[t] = (v8f){};

  // prologue: slab 0 -> buffer set A
  v16bf aA = pack_frag(arow, arow + 8);
  v16bf bA[NT];
#pragma unroll
  for (int t = 0; t < NT; ++t) bA[t] = pack_frag(brow[t], brow[t] + 8);

  // ping-pong over pairs of 32-deep K slabs (K % 64 == 0)
  for (int kb = 0; kb < K; kb += 64) {
    // load slab kb+32 -> buffer set B (always exists: slab count is even)
    v16bf aB = pack_frag(arow + kb + 32, arow + kb + 40);
    v16bf bB[NT];
#pragma unroll
    for (int t = 0; t < NT; ++t)
      bB[t] = pack_frag(brow[t] + kb + 32, brow[t] + kb + 40);

    // drain slab kb from buffer set A
#pragma unroll
    for (int t = 0; t < NT; ++t)
      acc[t] = __builtin_amdgcn_wmma_f32_16x16x32_bf16(
          false, aA, false, bA[t], (short)0, acc[t], false, false);

    // load slab kb+64 -> buffer set A (skip past the end)
    if (kb + 64 < K) {
      aA = pack_frag(arow + kb + 64, arow + kb + 72);
#pragma unroll
      for (int t = 0; t < NT; ++t)
        bA[t] = pack_frag(brow[t] + kb + 64, brow[t] + kb + 72);
    }

    // drain slab kb+32 from buffer set B
#pragma unroll
    for (int t = 0; t < NT; ++t)
      acc[t] = __builtin_amdgcn_wmma_f32_16x16x32_bf16(
          false, aB, false, bB[t], (short)0, acc[t], false, false);
  }

  int mrow = tm + (hsel << 3);
#pragma unroll
  for (int t = 0; t < NT; ++t) {
    int ccol = tn + (t << 4) + r;
    float badd = bias ? bias[ccol] : 0.0f;
#pragma unroll
    for (int g = 0; g < 8; ++g)
      C[(size_t)(mrow + g) * N + ccol] = acc[t][g] + badd;
  }
}

// ---------------------------------------------------------------------------
// Gathered attention. One wave per (query, head). Lane L owns neighbors
// {L, L+32} for scoring and output dims {L, L+32} for accumulation.
// qkv layout per token: [ q(16x64) | k(16x64) | v(16x64) ] = 3072 floats.
// Writes softmax(qK^T * scale) @ V as bf16 (feeds the WMMA output GEMM).
// ---------------------------------------------------------------------------
#define ATT_WAVES 8
__global__ __launch_bounds__(256) void attn_kernel(
    const float* __restrict__ qkv, const int* __restrict__ routes,
    unsigned short* __restrict__ attn_bf, int pairs) {
  __shared__ float lds_q[ATT_WAVES][HEAD_DIM];
  __shared__ float lds_p[ATT_WAVES][K_NEIGH];
  __shared__ int   lds_nbr[ATT_WAVES][K_NEIGH];

  int wave = threadIdx.x >> 5;
  int lane = threadIdx.x & 31;
  int pair = blockIdx.x * ATT_WAVES + wave;
  if (pair >= pairs) return;
  int qi = pair >> 4;    // token index
  int h  = pair & 15;    // head

  const size_t rowstride = 3 * DIM;
  const float* qrow = qkv + (size_t)qi * rowstride + h * HEAD_DIM;

  lds_q[wave][lane]      = qrow[lane];
  lds_q[wave][lane + 32] = qrow[lane + 32];
  int nb0 = routes[qi * K_NEIGH + lane];
  int nb1 = routes[qi * K_NEIGH + lane + 32];
  lds_nbr[wave][lane]      = nb0;
  lds_nbr[wave][lane + 32] = nb1;

  // scores: lane's two neighbors, vectorized float4 over head dim
  const float4* k0 = reinterpret_cast<const float4*>(
      qkv + (size_t)nb0 * rowstride + DIM + h * HEAD_DIM);
  const float4* k1 = reinterpret_cast<const float4*>(
      qkv + (size_t)nb1 * rowstride + DIM + h * HEAD_DIM);
  const float4* qv = reinterpret_cast<const float4*>(&lds_q[wave][0]);
  float s0 = 0.0f, s1 = 0.0f;
#pragma unroll
  for (int d = 0; d < HEAD_DIM / 4; ++d) {
    float4 qq = qv[d];
    float4 a = k0[d];
    float4 b = k1[d];
    s0 += qq.x * a.x + qq.y * a.y + qq.z * a.z + qq.w * a.w;
    s1 += qq.x * b.x + qq.y * b.y + qq.z * b.z + qq.w * b.w;
  }
  s0 *= ATTN_SCALE;
  s1 *= ATTN_SCALE;
  lds_p[wave][lane]      = s0;
  lds_p[wave][lane + 32] = s1;

  // wave-local softmax (lockstep wave32: LDS ops from one wave stay in order)
  float m = -3.0e38f;
  for (int j = 0; j < K_NEIGH; ++j) m = fmaxf(m, lds_p[wave][j]);
  float e0 = __expf(s0 - m);
  float e1 = __expf(s1 - m);
  lds_p[wave][lane]      = e0;
  lds_p[wave][lane + 32] = e1;
  float sum = 0.0f;
  for (int j = 0; j < K_NEIGH; ++j) sum += lds_p[wave][j];
  float inv = 1.0f / sum;

  // output: lane owns dims {lane, lane+32}; v-row reads are lane-contiguous
  float o0 = 0.0f, o1 = 0.0f;
  for (int j = 0; j < K_NEIGH; ++j) {
    float p = lds_p[wave][j];
    const float* vrow =
        qkv + (size_t)lds_nbr[wave][j] * rowstride + 2 * DIM + h * HEAD_DIM;
    o0 += p * vrow[lane];
    o1 += p * vrow[lane + 32];
  }
  o0 *= inv;
  o1 *= inv;
  size_t obase = (size_t)qi * DIM + h * HEAD_DIM;
  attn_bf[obase + lane]      = f32_to_bf16_rte(o0);
  attn_bf[obase + lane + 32] = f32_to_bf16_rte(o1);
}

// ---------------------------------------------------------------------------
// Host-side orchestration
// ---------------------------------------------------------------------------
extern "C" void kernel_launch(void* const* d_in, const int* in_sizes, int n_in,
                              void* d_out, int out_size, void* d_ws, size_t ws_size,
                              hipStream_t stream) {
  const float* x      = (const float*)d_in[0];
  const float* cantor = (const float*)d_in[1];
  const float* W_qkv  = (const float*)d_in[2];
  const float* b_qkv  = (const float*)d_in[3];
  const float* W_out  = (const float*)d_in[4];
  const float* b_out  = (const float*)d_in[5];
  float* out = (float*)d_out;
  const int n = in_sizes[1];  // number of tokens (4096)

  char* ws = (char*)d_ws;
  size_t off = 0;
  auto wsalloc = [&](size_t bytes) -> void* {
    void* p = (void*)(ws + off);
    off = (off + bytes + 255) & ~(size_t)255;
    return p;
  };
  unsigned short* x_bf    = (unsigned short*)wsalloc((size_t)n * DIM * 2);
  unsigned short* wqkvT   = (unsigned short*)wsalloc((size_t)3 * DIM * DIM * 2);
  unsigned short* woutT   = (unsigned short*)wsalloc((size_t)DIM * DIM * 2);
  float*          qkv     = (float*)wsalloc((size_t)n * 3 * DIM * 4);
  int*            routes  = (int*)wsalloc((size_t)n * K_NEIGH * 4);
  unsigned short* attn_bf = (unsigned short*)wsalloc((size_t)n * DIM * 2);

  // 1) precision conversion / weight pre-transpose (one-time O(bytes) passes)
  {
    int cnt = n * DIM;
    cvt_f32_bf16<<<(cnt + 255) / 256, 256, 0, stream>>>(x, x_bf, cnt);
  }
  {
    int cnt = DIM * 3 * DIM;
    transpose_cvt_f32_bf16<<<(cnt + 255) / 256, 256, 0, stream>>>(
        W_qkv, wqkvT, DIM, 3 * DIM);
  }
  {
    int cnt = DIM * DIM;
    transpose_cvt_f32_bf16<<<(cnt + 255) / 256, 256, 0, stream>>>(
        W_out, woutT, DIM, DIM);
  }

  // 2) neighbor routing (top-64 in 1-D coordinate space)
  routes_kernel<<<(n + 63) / 64, 64, 0, stream>>>(cantor, routes, n);

  // 3) QKV projection: (n x 1024) @ (1024 x 3072) + b_qkv  [WMMA bf16]
  {
    int strips = (n >> 4) * ((3 * DIM) >> 6);
    int blocks = (strips * 32 + 255) / 256;
    gemm_bf16_wmma<<<blocks, 256, 0, stream>>>(x_bf, wqkvT, b_qkv, qkv,
                                               n, 3 * DIM, DIM);
  }

  // 4) gathered attention (one wave per query-head pair)
  {
    int pairs = n * HEADS;
    attn_kernel<<<(pairs + ATT_WAVES - 1) / ATT_WAVES, 256, 0, stream>>>(
        qkv, routes, attn_bf, pairs);
  }

  // 5) output projection: (n x 1024) @ (1024 x 1024) + b_out  [WMMA bf16]
  {
    int strips = (n >> 4) * (DIM >> 6);
    int blocks = (strips * 32 + 255) / 256;
    gemm_bf16_wmma<<<blocks, 256, 0, stream>>>(attn_bf, woutT, b_out, out,
                                               n, DIM, DIM);
  }
}
